// GAT_2327872274792
// MI455X (gfx1250) — compile-verified, matrix operand order checked by reading, not run on hardware
//
#include <hip/hip_runtime.h>

// ---------------- types ----------------
typedef __bf16 bf16;
typedef __attribute__((ext_vector_type(16))) __bf16 v16bf;
typedef __attribute__((ext_vector_type(4)))  __bf16 v4bf;
typedef __attribute__((ext_vector_type(8)))  float  v8f;
typedef __attribute__((ext_vector_type(4)))  float  f32x4;
typedef __attribute__((ext_vector_type(4)))  unsigned int u32x4;

#define NN   4096
#define FF   512
#define HH   8
#define DD   64
#define GAT_ALPHA 0.2f
#define NEGINF (-9.0e15f)

union ABTile { u32x4 u[2]; v16bf v; };

// ---------------- f32 -> bf16 convert ----------------
__global__ void k_cvt(const float* __restrict__ s, bf16* __restrict__ d, int n) {
  int base = (blockIdx.x * blockDim.x + threadIdx.x) * 4;
  if (base >= n) return;
  f32x4 v = *(const f32x4*)(s + base);
  v4bf o;
  o[0] = (bf16)v[0]; o[1] = (bf16)v[1]; o[2] = (bf16)v[2]; o[3] = (bf16)v[3];
  *(v4bf*)(d + base) = o;
}

// ---------------- pack a K x C (row-major) matrix into WMMA-B tile layout ----
// Tile (kt,nt) covers K rows [kt*32,kt*32+32), cols [nt*16,nt*16+16).
// dst element tid = ((kt*(C/16)+nt)*32 + lane)*16 + e holds
//   B(kt*32 + (e&7) + 16*(e>>3) + 8*(lane>>4), nt*16 + (lane&15))
template <typename T>
__global__ void k_packB(const T* __restrict__ src, bf16* __restrict__ dst,
                        int K, int C, long srcHeadStride, long dstHeadStride) {
  long h = blockIdx.y;
  src += h * srcHeadStride;
  dst += h * dstHeadStride;
  int tid = blockIdx.x * blockDim.x + threadIdx.x;
  if (tid >= K * C) return;
  int e    = tid & 15;
  int lane = (tid >> 4) & 31;
  int tile = tid >> 9;
  int ntn  = C >> 4;
  int nt   = tile % ntn;
  int kt   = tile / ntn;
  int k = kt * 32 + (e & 7) + ((e >> 3) << 4) + ((lane >> 4) << 3);
  int c = nt * 16 + (lane & 15);
  dst[tid] = (bf16)(float)src[(long)k * C + c];
}

// ---------------- GEMM: Out[R,64] = A[R,K](bf16,row-major) x Bpack[K,64] ------
// one wave per 16-row tile; 4 WMMA accumulators cover N=64
__global__ void __launch_bounds__(32)
k_gemm(const bf16* __restrict__ A, const bf16* __restrict__ Bpack,
       bf16* __restrict__ Out, int K, long bHeadStride, long outHeadStride) {
  const int lane = threadIdx.x;
  const long h  = blockIdx.y;
  const int i0  = blockIdx.x * 16;
  Bpack += h * bHeadStride;
  Out   += h * outHeadStride;
  const int koff = (lane >> 4) << 3;
  const bf16* Arow = A + (long)(i0 + (lane & 15)) * K;

  v8f acc0 = {}, acc1 = {}, acc2 = {}, acc3 = {};
  for (int k0 = 0; k0 < K; k0 += 32) {
    ABTile a;
    a.u[0] = *(const u32x4*)(Arow + k0 + koff);
    a.u[1] = *(const u32x4*)(Arow + k0 + koff + 16);
    const bf16* bt = Bpack + ((long)(k0 >> 5) * 4) * 512 + lane * 16;
    ABTile b0, b1, b2, b3;
    b0.u[0] = *(const u32x4*)(bt);          b0.u[1] = *(const u32x4*)(bt + 8);
    b1.u[0] = *(const u32x4*)(bt + 512);    b1.u[1] = *(const u32x4*)(bt + 520);
    b2.u[0] = *(const u32x4*)(bt + 1024);   b2.u[1] = *(const u32x4*)(bt + 1032);
    b3.u[0] = *(const u32x4*)(bt + 1536);   b3.u[1] = *(const u32x4*)(bt + 1544);
    acc0 = __builtin_amdgcn_wmma_f32_16x16x32_bf16(false, a.v, false, b0.v, (short)0, acc0, false, false);
    acc1 = __builtin_amdgcn_wmma_f32_16x16x32_bf16(false, a.v, false, b1.v, (short)0, acc1, false, false);
    acc2 = __builtin_amdgcn_wmma_f32_16x16x32_bf16(false, a.v, false, b2.v, (short)0, acc2, false, false);
    acc3 = __builtin_amdgcn_wmma_f32_16x16x32_bf16(false, a.v, false, b3.v, (short)0, acc3, false, false);
  }
  // C layout: row = i0 + v + 8*(lane>>4), col = nt*16 + (lane&15)
  const int r0 = i0 + koff;
  const int cc = lane & 15;
#pragma unroll
  for (int v = 0; v < 8; ++v) {
    long r = r0 + v;
    Out[r * 64 + cc]      = (bf16)acc0[v];
    Out[r * 64 + 16 + cc] = (bf16)acc1[v];
    Out[r * 64 + 32 + cc] = (bf16)acc2[v];
    Out[r * 64 + 48 + cc] = (bf16)acc3[v];
  }
}

// ---------------- f_src / f_dst: 64-length dots of Wh rows with a ------------
__global__ void k_fsd(const bf16* __restrict__ Wh, const float* __restrict__ a,
                      float* __restrict__ fs, float* __restrict__ fd, int n_per_head) {
  int t = blockIdx.x * blockDim.x + threadIdx.x;
  int h = t / n_per_head;
  const bf16* w = Wh + (long)t * 64;
  const float* ah = a + (long)h * 128;
  float s = 0.f, d = 0.f;
#pragma unroll 8
  for (int k = 0; k < 64; ++k) {
    float wv = (float)w[k];
    s += wv * ah[k];
    d += wv * ah[64 + k];
  }
  fs[t] = s;
  fd[t] = d;
}

// ---------------- softmax row statistics (max + sum) -------------------------
// one wave per (head,row); adjacency row stays L2-resident across passes
__global__ void k_stats(const float* __restrict__ adj, const float* __restrict__ fs,
                        const float* __restrict__ fd, float* __restrict__ m,
                        float* __restrict__ l) {
  const int lane = threadIdx.x & 31;
  const int wv   = threadIdx.x >> 5;
  const long h   = blockIdx.y;
  const int i    = blockIdx.x * 8 + wv;
  const float* arow = adj + (long)i * NN;
  const float* fdh  = fd + h * NN;
  const float fsi = fs[h * NN + i];

  float mx = -3.0e38f;
  for (int j = lane * 4; j < NN; j += 128) {
    f32x4 av = *(const f32x4*)(arow + j);
    f32x4 dv = *(const f32x4*)(fdh + j);
#pragma unroll
    for (int q = 0; q < 4; ++q) {
      float e = fsi + dv[q];
      e = e > 0.f ? e : GAT_ALPHA * e;
      e = av[q] > 0.f ? e : NEGINF;
      mx = fmaxf(mx, e);
    }
  }
#pragma unroll
  for (int o = 16; o > 0; o >>= 1) mx = fmaxf(mx, __shfl_xor(mx, o, 32));

  float sum = 0.f;
  for (int j = lane * 4; j < NN; j += 128) {
    f32x4 av = *(const f32x4*)(arow + j);
    f32x4 dv = *(const f32x4*)(fdh + j);
#pragma unroll
    for (int q = 0; q < 4; ++q) {
      float e = fsi + dv[q];
      e = e > 0.f ? e : GAT_ALPHA * e;
      e = av[q] > 0.f ? e : NEGINF;
      sum += __expf(e - mx);
    }
  }
#pragma unroll
  for (int o = 16; o > 0; o >>= 1) sum += __shfl_xor(sum, o, 32);

  if (lane == 0) { m[h * NN + i] = mx; l[h * NN + i] = sum; }
}

// ---------------- fused softmax(e) @ Wh, never materializing attn ------------
// MODE 0: ELU -> bf16, concat into h1b[N, H*64]   MODE 1: ELU -> f32 d_out[N,64]
template <int MODE>
__global__ void __launch_bounds__(32)
k_attn(const float* __restrict__ adj, const float* __restrict__ fs,
       const float* __restrict__ fd, const float* __restrict__ m,
       const float* __restrict__ l, const bf16* __restrict__ WhPack,
       void* __restrict__ outp) {
  const int lane = threadIdx.x;
  const long h  = blockIdx.y;
  const int i0  = blockIdx.x * 16;
  const int i   = i0 + (lane & 15);
  const int koff = (lane >> 4) << 3;
  const float* arow = adj + (long)i * NN;
  const float* fdh  = fd + h * NN;
  const bf16*  BP   = WhPack + h * ((long)NN * 64);
  const float fsi  = fs[h * NN + i];
  const float mi   = m[h * NN + i];
  const float invl = 1.0f / l[h * NN + i];

  v8f acc0 = {}, acc1 = {}, acc2 = {}, acc3 = {};
  for (int j0 = 0; j0 < NN; j0 += 32) {
    const int jb = j0 + koff;
    f32x4 a0 = *(const f32x4*)(arow + jb);
    f32x4 a1 = *(const f32x4*)(arow + jb + 4);
    f32x4 a2 = *(const f32x4*)(arow + jb + 16);
    f32x4 a3 = *(const f32x4*)(arow + jb + 20);
    f32x4 d0 = *(const f32x4*)(fdh + jb);
    f32x4 d1 = *(const f32x4*)(fdh + jb + 4);
    f32x4 d2 = *(const f32x4*)(fdh + jb + 16);
    f32x4 d3 = *(const f32x4*)(fdh + jb + 20);
    float av[16], dv[16];
    *(f32x4*)&av[0] = a0; *(f32x4*)&av[4] = a1; *(f32x4*)&av[8] = a2; *(f32x4*)&av[12] = a3;
    *(f32x4*)&dv[0] = d0; *(f32x4*)&dv[4] = d1; *(f32x4*)&dv[8] = d2; *(f32x4*)&dv[12] = d3;

    // P computed directly in WMMA A layout: lane row i, K element e -> column
    // jb + (e&7) + 16*(e>>3), which is exactly av[e]/dv[e].
    v16bf A;
#pragma unroll
    for (int e = 0; e < 16; ++e) {
      float ev = fsi + dv[e];
      ev = ev > 0.f ? ev : GAT_ALPHA * ev;
      ev = av[e] > 0.f ? ev : NEGINF;
      A[e] = (bf16)(__expf(ev - mi) * invl);
    }

    const bf16* bt = BP + ((long)(j0 >> 5) * 4) * 512 + lane * 16;
    ABTile b0, b1, b2, b3;
    b0.u[0] = *(const u32x4*)(bt);          b0.u[1] = *(const u32x4*)(bt + 8);
    b1.u[0] = *(const u32x4*)(bt + 512);    b1.u[1] = *(const u32x4*)(bt + 520);
    b2.u[0] = *(const u32x4*)(bt + 1024);   b2.u[1] = *(const u32x4*)(bt + 1032);
    b3.u[0] = *(const u32x4*)(bt + 1536);   b3.u[1] = *(const u32x4*)(bt + 1544);
    acc0 = __builtin_amdgcn_wmma_f32_16x16x32_bf16(false, A, false, b0.v, (short)0, acc0, false, false);
    acc1 = __builtin_amdgcn_wmma_f32_16x16x32_bf16(false, A, false, b1.v, (short)0, acc1, false, false);
    acc2 = __builtin_amdgcn_wmma_f32_16x16x32_bf16(false, A, false, b2.v, (short)0, acc2, false, false);
    acc3 = __builtin_amdgcn_wmma_f32_16x16x32_bf16(false, A, false, b3.v, (short)0, acc3, false, false);
  }

  const int r0 = i0 + koff;
  const int cc = lane & 15;
  if (MODE == 0) {
    bf16* out = (bf16*)outp;
#pragma unroll
    for (int v = 0; v < 8; ++v) {
      long r = r0 + v;
      float c0 = acc0[v], c1 = acc1[v], c2 = acc2[v], c3 = acc3[v];
      c0 = c0 > 0.f ? c0 : (__expf(c0) - 1.f);
      c1 = c1 > 0.f ? c1 : (__expf(c1) - 1.f);
      c2 = c2 > 0.f ? c2 : (__expf(c2) - 1.f);
      c3 = c3 > 0.f ? c3 : (__expf(c3) - 1.f);
      out[r * 512 + h * 64 + cc]      = (bf16)c0;
      out[r * 512 + h * 64 + 16 + cc] = (bf16)c1;
      out[r * 512 + h * 64 + 32 + cc] = (bf16)c2;
      out[r * 512 + h * 64 + 48 + cc] = (bf16)c3;
    }
  } else {
    float* out = (float*)outp;
#pragma unroll
    for (int v = 0; v < 8; ++v) {
      long r = r0 + v;
      float c0 = acc0[v], c1 = acc1[v], c2 = acc2[v], c3 = acc3[v];
      out[r * 64 + cc]      = c0 > 0.f ? c0 : (__expf(c0) - 1.f);
      out[r * 64 + 16 + cc] = c1 > 0.f ? c1 : (__expf(c1) - 1.f);
      out[r * 64 + 32 + cc] = c2 > 0.f ? c2 : (__expf(c2) - 1.f);
      out[r * 64 + 48 + cc] = c3 > 0.f ? c3 : (__expf(c3) - 1.f);
    }
  }
}

// ---------------- host side ----------------
extern "C" void kernel_launch(void* const* d_in, const int* in_sizes, int n_in,
                              void* d_out, int out_size, void* d_ws, size_t ws_size,
                              hipStream_t stream) {
  const float* x   = (const float*)d_in[0];   // [N,F]
  const float* adj = (const float*)d_in[1];   // [N,N]
  const float* W   = (const float*)d_in[2];   // [H,F,D]
  const float* a   = (const float*)d_in[3];   // [H,2D]
  const float* Wo  = (const float*)d_in[4];   // [F,D]
  const float* ao  = (const float*)d_in[5];   // [2D]
  float* out = (float*)d_out;                 // [N,NCLASS] f32

  char* ws = (char*)d_ws;
  size_t off = 0;
  auto alloc = [&](size_t bytes) -> char* {
    char* p = ws + off;
    off += (bytes + 255) & ~(size_t)255;
    return p;
  };

  bf16* xb      = (bf16*)alloc((size_t)NN * FF * 2);
  bf16* Wpack   = (bf16*)alloc((size_t)HH * FF * DD * 2);
  bf16* WoPack  = (bf16*)alloc((size_t)FF * DD * 2);
  bf16* Whb     = (bf16*)alloc((size_t)HH * NN * DD * 2);
  bf16* WhPack  = (bf16*)alloc((size_t)HH * NN * DD * 2);
  float* fs1    = (float*)alloc((size_t)HH * NN * 4);
  float* fd1    = (float*)alloc((size_t)HH * NN * 4);
  float* m1     = (float*)alloc((size_t)HH * NN * 4);
  float* l1     = (float*)alloc((size_t)HH * NN * 4);
  bf16* h1b     = (bf16*)alloc((size_t)NN * FF * 2);
  bf16* Wh2     = (bf16*)alloc((size_t)NN * DD * 2);
  bf16* Wh2Pack = (bf16*)alloc((size_t)NN * DD * 2);
  float* fs2    = (float*)alloc((size_t)NN * 4);
  float* fd2    = (float*)alloc((size_t)NN * 4);
  float* m2     = (float*)alloc((size_t)NN * 4);
  float* l2     = (float*)alloc((size_t)NN * 4);
  (void)ws_size; (void)in_sizes; (void)n_in; (void)out_size;

  // ---- layer 1 ----
  k_cvt<<<(NN * FF) / 1024, 256, 0, stream>>>(x, xb, NN * FF);
  k_packB<float><<<dim3((FF * DD) / 256, HH), 256, 0, stream>>>(
      W, Wpack, FF, DD, (long)FF * DD, (long)FF * DD);
  k_packB<float><<<dim3((FF * DD) / 256, 1), 256, 0, stream>>>(
      Wo, WoPack, FF, DD, 0, 0);
  k_gemm<<<dim3(NN / 16, HH), 32, 0, stream>>>(
      xb, Wpack, Whb, FF, (long)FF * DD, (long)NN * DD);
  k_packB<bf16><<<dim3((NN * DD) / 256, HH), 256, 0, stream>>>(
      Whb, WhPack, NN, DD, (long)NN * DD, (long)NN * DD);
  k_fsd<<<(HH * NN) / 256, 256, 0, stream>>>(Whb, a, fs1, fd1, NN);
  k_stats<<<dim3(NN / 8, HH), 256, 0, stream>>>(adj, fs1, fd1, m1, l1);
  k_attn<0><<<dim3(NN / 16, HH), 32, 0, stream>>>(adj, fs1, fd1, m1, l1, WhPack, h1b);

  // ---- layer 2 ----
  k_gemm<<<dim3(NN / 16, 1), 32, 0, stream>>>(h1b, WoPack, Wh2, FF, 0, 0);
  k_packB<bf16><<<dim3((NN * DD) / 256, 1), 256, 0, stream>>>(Wh2, Wh2Pack, NN, DD, 0, 0);
  k_fsd<<<NN / 256, 256, 0, stream>>>(Wh2, ao, fs2, fd2, NN);
  k_stats<<<dim3(NN / 8, 1), 256, 0, stream>>>(adj, fs2, fd2, m2, l2);
  k_attn<1><<<dim3(NN / 16, 1), 32, 0, stream>>>(adj, fs2, fd2, m2, l2, Wh2Pack, out);
}
